// BertCRFMultiTaskModel_87110526697657
// MI455X (gfx1250) — matrix operand order repaired and checked by reading, not compiled, not run on hardware
//
#include <hip/hip_runtime.h>
#include <stdint.h>

// ---------------- problem constants ----------------
#define B_   64
#define S_   256
#define H_   768
#define L_   12
#define NH_  12
#define DH_  64
#define FF_  3072
#define NI_  10
#define NS_  64
#define BS_  (B_ * S_)

typedef __attribute__((ext_vector_type(16))) _Float16 v16h;
typedef __attribute__((ext_vector_type(8)))  float    v8f;

// ---------------- gfx1250 async LDS copy (guarded) ----------------
#if defined(__AMDGCN__) && __has_builtin(__builtin_amdgcn_global_load_async_to_lds_b128)
#define HAVE_ASYNC_LDS 1
typedef int async_v4i __attribute__((vector_size(16)));
typedef __attribute__((address_space(1))) async_v4i g_v4i;
typedef __attribute__((address_space(3))) async_v4i l_v4i;
__device__ __forceinline__ void cp16_to_lds(const _Float16* g, _Float16* l) {
    __builtin_amdgcn_global_load_async_to_lds_b128((g_v4i*)(g), (l_v4i*)(l), 0, 0);
}
#if __has_builtin(__builtin_amdgcn_s_wait_asynccnt)
#define WAIT_ASYNC(n) __builtin_amdgcn_s_wait_asynccnt(n)
#else
#define WAIT_ASYNC(n) asm volatile("s_wait_asynccnt %0" ::"i"(n) : "memory")
#endif
#else
#define HAVE_ASYNC_LDS 0
__device__ __forceinline__ void cp16_to_lds(const _Float16* g, _Float16* l) {
    *(uint4*)l = *(const uint4*)g;
}
#define WAIT_ASYNC(n) ((void)0)
#endif

// ================= transpose + fp32->f16 convert =================
// dst[bt][n][k] = (f16) src[bt][k][n]
__global__ void tconv_kernel(const float* __restrict__ src, _Float16* __restrict__ dst,
                             int batch, int K, int N) {
    long idx = (long)blockIdx.x * blockDim.x + threadIdx.x;
    long total = (long)batch * K * N;
    if (idx >= total) return;
    long bt  = idx / ((long)K * N);
    long rem = idx % ((long)K * N);
    long n = rem / K;
    long k = rem % K;
    dst[idx] = (_Float16)src[bt * (long)K * N + k * (long)N + n];
}

// ================= V transpose: [b,s,h,d] -> [b,h,d,s] (f16) =================
__global__ void vtrans_kernel(const _Float16* __restrict__ vh, _Float16* __restrict__ vt) {
    long idx = (long)blockIdx.x * blockDim.x + threadIdx.x;   // over B*NH*DH*S
    long s  = idx % S_;
    long r  = idx / S_;
    long d  = r % DH_;
    long bh = r / DH_;
    long h  = bh % NH_;
    long b  = bh / NH_;
    vt[idx] = vh[(b * S_ + s) * H_ + h * DH_ + d];
}

// ================= embeddings (sum of three tables) =================
__global__ void embed_kernel(const int* __restrict__ ids, const int* __restrict__ tt,
                             const float* __restrict__ we, const float* __restrict__ pe,
                             const float* __restrict__ te, float* __restrict__ x) {
    int row = blockIdx.x;              // b*S + s
    int s   = row % S_;
    int id  = ids[row];
    int t   = tt[row];
    for (int i = threadIdx.x; i < H_; i += blockDim.x)
        x[(long)row * H_ + i] = we[(long)id * H_ + i] + pe[(long)s * H_ + i] + te[(long)t * H_ + i];
}

// ================= (residual add) + LayerNorm, dual f32/f16 output =================
__launch_bounds__(256)
__global__ void add_ln_kernel(const float* __restrict__ base,   // may be null
                              const float* __restrict__ t,
                              const float* __restrict__ g, const float* __restrict__ bta,
                              float* __restrict__ xout, _Float16* __restrict__ hout) {
    __shared__ float red[16];
    long row = blockIdx.x;
    const float* tp = t + row * H_;
    const float* bp = base ? base + row * H_ : nullptr;
    float v[3];
    float s = 0.f, s2 = 0.f;
#pragma unroll
    for (int j = 0; j < 3; ++j) {
        int i = threadIdx.x + j * 256;
        float x = tp[i] + (bp ? bp[i] : 0.f);
        v[j] = x; s += x; s2 += x * x;
    }
#pragma unroll
    for (int o = 16; o > 0; o >>= 1) { s += __shfl_xor(s, o, 32); s2 += __shfl_xor(s2, o, 32); }
    int wave = threadIdx.x >> 5, lane = threadIdx.x & 31;
    if (lane == 0) { red[wave] = s; red[8 + wave] = s2; }
    __syncthreads();
    if (threadIdx.x == 0) {
        float a = 0.f, b2 = 0.f;
        for (int w = 0; w < 8; ++w) { a += red[w]; b2 += red[8 + w]; }
        red[0] = a; red[8] = b2;
    }
    __syncthreads();
    float mean = red[0] * (1.f / H_);
    float var  = red[8] * (1.f / H_) - mean * mean;
    float rstd = rsqrtf(var + 1e-12f);
#pragma unroll
    for (int j = 0; j < 3; ++j) {
        int i = threadIdx.x + j * 256;
        float y = (v[j] - mean) * rstd * g[i] + bta[i];
        xout[row * H_ + i] = y;
        hout[row * H_ + i] = (_Float16)y;
    }
}

// ================= WMMA GEMM =================
// C[M,N] = A[M,K] * B^T + bias (B stored [N][K] row-major), optional exact GELU.
// Batched via blockIdx.z -> (outer, inner). Requires M%128==0, N%64==0, K%32==0.
#define BM 128
#define BN 64
#define BKT 32
#define TS  (BKT + 8)   // padded LDS row stride (halves)

__launch_bounds__(256)
__global__ void wmma_gemm_kernel(const _Float16* __restrict__ A, long lda, long sAin, long sAout,
                                 const _Float16* __restrict__ Bm, long ldb, long sBin, long sBout,
                                 float* __restrict__ outF, _Float16* __restrict__ outH,
                                 long ldo, long sOin, long sOout,
                                 const float* __restrict__ bias, int act,
                                 int K, int innerCount) {
    __shared__ __align__(16) _Float16 sA[2][BM * TS];
    __shared__ __align__(16) _Float16 sB[2][BN * TS];

    int bz = blockIdx.z;
    int bi = bz % innerCount, bo = bz / innerCount;
    const _Float16* Ab = A  + (long)bo * sAout + (long)bi * sAin;
    const _Float16* Bb = Bm + (long)bo * sBout + (long)bi * sBin;
    long obase = (long)bo * sOout + (long)bi * sOin;

    int rowBase = blockIdx.y * BM;
    int colBase = blockIdx.x * BN;

    int tid  = threadIdx.x;
    int wave = tid >> 5;
    int lane = tid & 31;
    int wm = wave >> 1;        // 0..3 : 32-row slab
    int wn = wave & 1;         // 0..1 : 32-col slab

    v8f acc00 = {}, acc01 = {}, acc10 = {}, acc11 = {};

    // tile loaders: A 128x32 (2x16B per thread), B 64x32 (1x16B per thread)
    int ra = tid >> 1;
    int ca = (tid & 1) * 16;
    int rb = tid >> 2;
    int cb = (tid & 3) * 8;

    auto load_tiles = [&](int buf, int k0) {
        const _Float16* srcA = Ab + (long)(rowBase + ra) * lda + (k0 + ca);
        cp16_to_lds(srcA,     &sA[buf][ra * TS + ca]);
        cp16_to_lds(srcA + 8, &sA[buf][ra * TS + ca + 8]);
        cp16_to_lds(Bb + (long)(colBase + rb) * ldb + (k0 + cb), &sB[buf][rb * TS + cb]);
    };

    int nk = K / BKT;
    load_tiles(0, 0);
    for (int kt = 0; kt < nk; ++kt) {
        int cur = kt & 1;
        __syncthreads();                       // prior reads of next buffer done
        if (kt + 1 < nk) { load_tiles(1 - cur, (kt + 1) * BKT); WAIT_ASYNC(3); }
        else             { WAIT_ASYNC(0); }
        __syncthreads();                       // tile 'cur' resident in LDS

        const _Float16* tA = sA[cur];
        const _Float16* tB = sB[cur];

        // A fragments (contiguous k order: k = 16*(lane/16) + s, matching B)
        int arow = wm * 32 + (lane & 15);
        int akb  = (lane < 16) ? 0 : 16;
        v16h a0, a1;
#pragma unroll
        for (int v = 0; v < 8; ++v) {
            int kk = akb + 2 * v;
            a0[2 * v]     = tA[arow * TS + kk];
            a0[2 * v + 1] = tA[arow * TS + kk + 1];
            a1[2 * v]     = tA[(arow + 16) * TS + kk];
            a1[2 * v + 1] = tA[(arow + 16) * TS + kk + 1];
        }
        // B fragments (contiguous k order: k = 16*(lane/16) + s, matching A), from B^T tile [n][k]
        int bcol = wn * 32 + (lane & 15);
        int bkb  = (lane < 16) ? 0 : 16;
        v16h b0, b1f;
#pragma unroll
        for (int v = 0; v < 8; ++v) {
            int kk = bkb + 2 * v;
            b0[2 * v]      = tB[bcol * TS + kk];
            b0[2 * v + 1]  = tB[bcol * TS + kk + 1];
            b1f[2 * v]     = tB[(bcol + 16) * TS + kk];
            b1f[2 * v + 1] = tB[(bcol + 16) * TS + kk + 1];
        }
        acc00 = __builtin_amdgcn_wmma_f32_16x16x32_f16(false, a0, false, b0,  (short)0, acc00, false, false);
        acc01 = __builtin_amdgcn_wmma_f32_16x16x32_f16(false, a0, false, b1f, (short)0, acc01, false, false);
        acc10 = __builtin_amdgcn_wmma_f32_16x16x32_f16(false, a1, false, b0,  (short)0, acc10, false, false);
        acc11 = __builtin_amdgcn_wmma_f32_16x16x32_f16(false, a1, false, b1f, (short)0, acc11, false, false);
    }

    // epilogue: C/D layout -> lane col = lane&15 ; rows = ((lane<16)?0:8)+r
    int nc0 = colBase + wn * 32 + (lane & 15);
    int nc1 = nc0 + 16;
    int mb  = rowBase + wm * 32 + ((lane < 16) ? 0 : 8);
    float bias0 = bias ? bias[nc0] : 0.f;
    float bias1 = bias ? bias[nc1] : 0.f;
#pragma unroll
    for (int mi = 0; mi < 2; ++mi) {
        v8f aN0 = mi ? acc10 : acc00;
        v8f aN1 = mi ? acc11 : acc01;
#pragma unroll
        for (int r = 0; r < 8; ++r) {
            int m = mb + mi * 16 + r;
            float v0 = aN0[r] + bias0;
            float v1 = aN1[r] + bias1;
            if (act == 1) {
                v0 = 0.5f * v0 * (1.f + erff(v0 * 0.70710678118654752f));
                v1 = 0.5f * v1 * (1.f + erff(v1 * 0.70710678118654752f));
            }
            long o = obase + (long)m * ldo;
            if (outF) { outF[o + nc0] = v0; outF[o + nc1] = v1; }
            if (outH) { outH[o + nc0] = (_Float16)v0; outH[o + nc1] = (_Float16)v1; }
        }
    }
}

// ================= masked, scaled softmax over 256 keys (wave per row) =================
__launch_bounds__(256)
__global__ void softmax_kernel(_Float16* __restrict__ sc, const int* __restrict__ mask) {
    int wave = threadIdx.x >> 5, lane = threadIdx.x & 31;
    long row = (long)blockIdx.x * 8 + wave;          // < B*NH*S
    long b   = row / ((long)NH_ * S_);
    _Float16* p = sc + row * S_;
    float vals[8];
    float mx = -1e30f;
#pragma unroll
    for (int j = 0; j < 8; ++j) {
        int kpos = lane + j * 32;
        float m = (float)mask[b * S_ + kpos];
        float v = (float)p[kpos] * 0.125f + (1.f - m) * -1e9f;
        vals[j] = v; mx = fmaxf(mx, v);
    }
#pragma unroll
    for (int o = 16; o > 0; o >>= 1) mx = fmaxf(mx, __shfl_xor(mx, o, 32));
    float sum = 0.f;
#pragma unroll
    for (int j = 0; j < 8; ++j) { vals[j] = expf(vals[j] - mx); sum += vals[j]; }
#pragma unroll
    for (int o = 16; o > 0; o >>= 1) sum += __shfl_xor(sum, o, 32);
    float inv = 1.f / sum;
#pragma unroll
    for (int j = 0; j < 8; ++j) p[lane + j * 32] = (_Float16)(vals[j] * inv);
}

// ================= intent head: logits[b][j] (10 waves per block) =================
__global__ void intent_logits_kernel(const float* __restrict__ x, const float* __restrict__ Wi,
                                     const float* __restrict__ bi, float* __restrict__ out) {
    int b = blockIdx.x;
    int wave = threadIdx.x >> 5, lane = threadIdx.x & 31;
    const float* xr = x + (long)b * S_ * H_;         // token s=0
    float s = 0.f;
    for (int i = lane; i < H_; i += 32) s += xr[i] * Wi[(long)i * NI_ + wave];
#pragma unroll
    for (int o = 16; o > 0; o >>= 1) s += __shfl_xor(s, o, 32);
    if (lane == 0) out[b * NI_ + wave] = s + bi[wave];
}

__global__ void intent_nll_kernel(const float* __restrict__ logits, const int* __restrict__ labels,
                                  float* __restrict__ nll) {
    int b = threadIdx.x;
    if (b >= B_) return;
    const float* lr = logits + b * NI_;
    float m = -1e30f;
    for (int j = 0; j < NI_; ++j) m = fmaxf(m, lr[j]);
    float s = 0.f;
    for (int j = 0; j < NI_; ++j) s += expf(lr[j] - m);
    nll[b] = (m + logf(s)) - lr[labels[b]];
}

// ================= CRF: gold score + forward recursion (block per batch, 64 threads) =================
__launch_bounds__(64)
__global__ void crf_kernel(const float* __restrict__ em, const int* __restrict__ tags,
                           const int* __restrict__ mask, const float* __restrict__ startv,
                           const float* __restrict__ endv, const float* __restrict__ trans,
                           float* __restrict__ llh) {
    __shared__ float tr[NS_ * NS_];
    __shared__ float alpha[NS_];
    __shared__ float goldsc;
    int b = blockIdx.x, tid = threadIdx.x;
    for (int i = tid; i < NS_ * NS_; i += NS_) tr[i] = trans[i];
    const float* emr = em + (long)b * S_ * NS_;
    alpha[tid] = startv[tid] + emr[tid];
    if (tid == 0) {
        int t0 = tags[b * S_];
        float sc = startv[t0] + emr[t0];
        int len = 0;
        for (int s = 0; s < S_; ++s) len += mask[b * S_ + s];
        for (int s = 1; s < S_; ++s) {
            int pv = tags[b * S_ + s - 1], cu = tags[b * S_ + s];
            sc += (trans[pv * NS_ + cu] + emr[(long)s * NS_ + cu]) * (float)mask[b * S_ + s];
        }
        sc += endv[tags[b * S_ + len - 1]];
        goldsc = sc;
    }
    __syncthreads();
    for (int s = 1; s < S_; ++s) {
        float m = -1e30f;
        for (int i = 0; i < NS_; ++i) m = fmaxf(m, alpha[i] + tr[i * NS_ + tid]);
        float sm = 0.f;
        for (int i = 0; i < NS_; ++i) sm += expf(alpha[i] + tr[i * NS_ + tid] - m);
        float nv = m + logf(sm) + emr[(long)s * NS_ + tid];
        float newa = ((float)mask[b * S_ + s] > 0.f) ? nv : alpha[tid];
        __syncthreads();
        alpha[tid] = newa;
        __syncthreads();
    }
    if (tid == 0) {
        float m = -1e30f;
        for (int j = 0; j < NS_; ++j) m = fmaxf(m, alpha[j] + endv[j]);
        float sm = 0.f;
        for (int j = 0; j < NS_; ++j) sm += expf(alpha[j] + endv[j] - m);
        llh[b] = goldsc - (m + logf(sm));
    }
}

__global__ void combine_kernel(const float* __restrict__ nll, const float* __restrict__ llh,
                               float* __restrict__ out) {
    if (threadIdx.x == 0 && blockIdx.x == 0) {
        float a = 0.f, c = 0.f;
        for (int b = 0; b < B_; ++b) { a += nll[b]; c += llh[b]; }
        out[0] = a / B_ + 2.f * (-(c / B_));
    }
}

// ================= host-side orchestration =================
extern "C" void kernel_launch(void* const* d_in, const int* in_sizes, int n_in,
                              void* d_out, int out_size, void* d_ws, size_t ws_size,
                              hipStream_t stream) {
    (void)in_sizes; (void)n_in; (void)out_size; (void)ws_size;
    const int*   input_ids  = (const int*)d_in[0];
    const int*   attn_mask  = (const int*)d_in[1];
    const int*   token_type = (const int*)d_in[2];
    const int*   intent_lab = (const int*)d_in[3];
    const int*   slot_lab   = (const int*)d_in[4];
    const float* word_emb   = (const float*)d_in[5];
    const float* pos_emb    = (const float*)d_in[6];
    const float* type_emb   = (const float*)d_in[7];
    const float* emb_g      = (const float*)d_in[8];
    const float* emb_b      = (const float*)d_in[9];
    const float* Wq = (const float*)d_in[10]; const float* bq = (const float*)d_in[11];
    const float* Wk = (const float*)d_in[12]; const float* bk = (const float*)d_in[13];
    const float* Wv = (const float*)d_in[14]; const float* bv = (const float*)d_in[15];
    const float* Wo = (const float*)d_in[16]; const float* bo = (const float*)d_in[17];
    const float* ln1g = (const float*)d_in[18]; const float* ln1b = (const float*)d_in[19];
    const float* W1 = (const float*)d_in[20]; const float* b1 = (const float*)d_in[21];
    const float* W2 = (const float*)d_in[22]; const float* b2 = (const float*)d_in[23];
    const float* ln2g = (const float*)d_in[24]; const float* ln2b = (const float*)d_in[25];
    const float* Wi = (const float*)d_in[26]; const float* bi = (const float*)d_in[27];
    const float* Ws = (const float*)d_in[28]; const float* bs = (const float*)d_in[29];
    const float* crf_start = (const float*)d_in[30];
    const float* crf_end   = (const float*)d_in[31];
    const float* crf_trans = (const float*)d_in[32];
    float* out = (float*)d_out;

    // workspace carve-up (deterministic)
    char* ws = (char*)d_ws;
    size_t off = 0;
    auto carve = [&](size_t bytes) -> char* {
        char* p = ws + off;
        off += (bytes + 255) & ~(size_t)255;
        return p;
    };
    _Float16* wq_t = (_Float16*)carve((size_t)L_ * H_ * H_ * 2);
    _Float16* wk_t = (_Float16*)carve((size_t)L_ * H_ * H_ * 2);
    _Float16* wv_t = (_Float16*)carve((size_t)L_ * H_ * H_ * 2);
    _Float16* wo_t = (_Float16*)carve((size_t)L_ * H_ * H_ * 2);
    _Float16* w1_t = (_Float16*)carve((size_t)L_ * H_ * FF_ * 2);
    _Float16* w2_t = (_Float16*)carve((size_t)L_ * FF_ * H_ * 2);
    _Float16* ws_t = (_Float16*)carve((size_t)H_ * NS_ * 2);
    float*    x    = (float*)   carve((size_t)BS_ * H_ * 4);
    _Float16* xh   = (_Float16*)carve((size_t)BS_ * H_ * 2);
    float*    tmp  = (float*)   carve((size_t)BS_ * H_ * 4);
    _Float16* qh   = (_Float16*)carve((size_t)BS_ * H_ * 2);
    _Float16* kh   = (_Float16*)carve((size_t)BS_ * H_ * 2);
    _Float16* vh   = (_Float16*)carve((size_t)BS_ * H_ * 2);
    _Float16* vt   = (_Float16*)carve((size_t)BS_ * H_ * 2);     // [b,h,d,s]
    _Float16* scor = (_Float16*)carve((size_t)B_ * NH_ * S_ * S_ * 2);
    _Float16* ctxh = (_Float16*)carve((size_t)BS_ * H_ * 2);
    _Float16* ff1h = (_Float16*)carve((size_t)BS_ * FF_ * 2);
    float*    slot = (float*)   carve((size_t)BS_ * NS_ * 4);
    float*    ilog = (float*)   carve((size_t)B_ * NI_ * 4);
    float*    nll  = (float*)   carve((size_t)B_ * 4);
    float*    llh  = (float*)   carve((size_t)B_ * 4);

    auto ceilDiv = [](long a, long b) { return (int)((a + b - 1) / b); };

    // 1) weight convert + transpose to f16 [N][K]
    {
        long e;
        e = (long)L_ * H_ * H_;
        tconv_kernel<<<ceilDiv(e, 256), 256, 0, stream>>>(Wq, wq_t, L_, H_, H_);
        tconv_kernel<<<ceilDiv(e, 256), 256, 0, stream>>>(Wk, wk_t, L_, H_, H_);
        tconv_kernel<<<ceilDiv(e, 256), 256, 0, stream>>>(Wv, wv_t, L_, H_, H_);
        tconv_kernel<<<ceilDiv(e, 256), 256, 0, stream>>>(Wo, wo_t, L_, H_, H_);
        e = (long)L_ * H_ * FF_;
        tconv_kernel<<<ceilDiv(e, 256), 256, 0, stream>>>(W1, w1_t, L_, H_, FF_);
        tconv_kernel<<<ceilDiv(e, 256), 256, 0, stream>>>(W2, w2_t, L_, FF_, H_);
        e = (long)H_ * NS_;
        tconv_kernel<<<ceilDiv(e, 256), 256, 0, stream>>>(Ws, ws_t, 1, H_, NS_);
    }

    // 2) embeddings + LN
    embed_kernel<<<BS_, 256, 0, stream>>>(input_ids, token_type, word_emb, pos_emb, type_emb, x);
    add_ln_kernel<<<BS_, 256, 0, stream>>>(nullptr, x, emb_g, emb_b, x, xh);

    const long SH = (long)S_ * H_;   // per-batch activation stride
    const long SS = (long)S_ * S_;   // per-head score stride
    const long DS = (long)DH_ * S_;  // per-head vt stride

    // 3) encoder layers
    for (int l = 0; l < L_; ++l) {
        const _Float16* wq = wq_t + (long)l * H_ * H_;
        const _Float16* wk = wk_t + (long)l * H_ * H_;
        const _Float16* wv = wv_t + (long)l * H_ * H_;
        const _Float16* wo = wo_t + (long)l * H_ * H_;
        const _Float16* w1 = w1_t + (long)l * H_ * FF_;
        const _Float16* w2 = w2_t + (long)l * FF_ * H_;

        dim3 gHH(H_ / BN, BS_ / BM, 1);
        // Q,K,V projections (f16 outputs only)
        wmma_gemm_kernel<<<gHH, 256, 0, stream>>>(xh, H_, 0, 0, wq, H_, 0, 0,
            nullptr, qh, H_, 0, 0, bq + (long)l * H_, 0, H_, 1);
        wmma_gemm_kernel<<<gHH, 256, 0, stream>>>(xh, H_, 0, 0, wk, H_, 0, 0,
            nullptr, kh, H_, 0, 0, bk + (long)l * H_, 0, H_, 1);
        wmma_gemm_kernel<<<gHH, 256, 0, stream>>>(xh, H_, 0, 0, wv, H_, 0, 0,
            nullptr, vh, H_, 0, 0, bv + (long)l * H_, 0, H_, 1);

        // V transpose -> [b,h,d,s]
        vtrans_kernel<<<(BS_ * H_) / 256, 256, 0, stream>>>(vh, vt);

        // scores = Q K^T  (batched over B*NH)
        dim3 gSc(S_ / BN, S_ / BM, B_ * NH_);
        wmma_gemm_kernel<<<gSc, 256, 0, stream>>>(qh, H_, DH_, SH, kh, H_, DH_, SH,
            nullptr, scor, S_, SS, (long)NH_ * SS, nullptr, 0, DH_, NH_);

        // masked scaled softmax (in-place on scores)
        softmax_kernel<<<(B_ * NH_ * S_) / 8, 256, 0, stream>>>(scor, attn_mask);

        // ctx = P V   (B^T = vt slice [d][s])
        dim3 gCtx(DH_ / BN, S_ / BM, B_ * NH_);
        wmma_gemm_kernel<<<gCtx, 256, 0, stream>>>(scor, S_, SS, (long)NH_ * SS, vt, S_, DS, (long)NH_ * DS,
            nullptr, ctxh, H_, DH_, SH, nullptr, 0, S_, NH_);

        // attention output projection -> tmp (f32)
        wmma_gemm_kernel<<<gHH, 256, 0, stream>>>(ctxh, H_, 0, 0, wo, H_, 0, 0,
            tmp, nullptr, H_, 0, 0, bo + (long)l * H_, 0, H_, 1);
        add_ln_kernel<<<BS_, 256, 0, stream>>>(x, tmp, ln1g + (long)l * H_, ln1b + (long)l * H_, x, xh);

        // FFN
        dim3 gF1(FF_ / BN, BS_ / BM, 1);
        wmma_gemm_kernel<<<gF1, 256, 0, stream>>>(xh, H_, 0, 0, w1, H_, 0, 0,
            nullptr, ff1h, FF_, 0, 0, b1 + (long)l * FF_, 1 /*gelu*/, H_, 1);
        wmma_gemm_kernel<<<gHH, 256, 0, stream>>>(ff1h, FF_, 0, 0, w2, FF_, 0, 0,
            tmp, nullptr, H_, 0, 0, b2 + (long)l * H_, 0, FF_, 1);
        add_ln_kernel<<<BS_, 256, 0, stream>>>(x, tmp, ln2g + (long)l * H_, ln2b + (long)l * H_, x, xh);
    }

    // 4) slot logits (f32)
    {
        dim3 gS(NS_ / BN, BS_ / BM, 1);
        wmma_gemm_kernel<<<gS, 256, 0, stream>>>(xh, H_, 0, 0, ws_t, H_, 0, 0,
            slot, nullptr, NS_, 0, 0, bs, 0, H_, 1);
    }

    // 5) intent head + losses
    intent_logits_kernel<<<B_, 320, 0, stream>>>(x, Wi, bi, ilog);
    intent_nll_kernel<<<1, 64, 0, stream>>>(ilog, intent_lab, nll);
    crf_kernel<<<B_, 64, 0, stream>>>(slot, slot_lab, attn_mask, crf_start, crf_end, crf_trans, llh);
    combine_kernel<<<1, 32, 0, stream>>>(nll, llh, out);
}
